// CrossGATLayer_83889301226223
// MI455X (gfx1250) — matrix operand order repaired and checked by reading, not compiled, not run on hardware
//
#include <hip/hip_runtime.h>
#include <hip/hip_bf16.h>

// MI455X / gfx1250, wave32, WMMA bf16 16x16x32 with f32 accumulation.

typedef __bf16 bf16_t;
typedef __bf16 v16bf __attribute__((ext_vector_type(16)));
typedef __bf16 v8bf  __attribute__((ext_vector_type(8)));
typedef __bf16 v4bf  __attribute__((ext_vector_type(4)));
typedef float  v8f   __attribute__((ext_vector_type(8)));

#define BS   16
#define T    1024
#define N2   2048          // 2*t nodes per batch
#define F    512           // IN_F == OUT_F
#define NROW (BS * N2)     // 32768 global rows
#define GAT_ALPHA 0.1f
#define GAT_NEG_INF (-9.0e15f)

__device__ __forceinline__ v16bf mk16(v8bf lo, v8bf hi) {
  v16bf r;
#pragma unroll
  for (int i = 0; i < 8; ++i) { r[i] = lo[i]; r[i + 8] = hi[i]; }
  return r;
}

// ---- Kernel 0: W (f32, row-major [k][n]) -> Wt (bf16, [n][k]) ------------
__global__ __launch_bounds__(256) void k_wprep(const float* __restrict__ W,
                                               bf16_t* __restrict__ Wt) {
  int idx = blockIdx.x * 256 + threadIdx.x;   // idx = n*F + k
  int n = idx >> 9, k = idx & (F - 1);
  Wt[idx] = (bf16_t)W[(size_t)k * F + n];
}

// ---- Kernel 1: Wh = concat(x_a,x_v) @ W  (bf16 WMMA, f32 accum) ----------
// Block: 256 thr = 8 waves; block covers 16 rows x 512 cols (wave: 16x64).
__global__ __launch_bounds__(256) void k_gemm_wh(const float* __restrict__ xa,
                                                 const float* __restrict__ xv,
                                                 const bf16_t* __restrict__ Wt,
                                                 bf16_t* __restrict__ Whbf,
                                                 bf16_t* __restrict__ WhbfT) {
  const int rt   = blockIdx.x;           // 128 row-tiles per batch
  const int wave = threadIdx.x >> 5;
  const int lane = threadIdx.x & 31;
  const int b    = rt >> 7;
  const int i0   = (rt & 127) << 4;      // row offset within batch
  const int lr   = lane & 15;
  const int hi   = lane >> 4;            // lane half

  const int pos = i0 + lr;               // this lane's A-matrix row
  const float* hrow = (pos < T) ? xa + ((size_t)b * T + pos) * F
                                : xv + ((size_t)b * T + (pos - T)) * F;
  const int n0 = wave * 64;

  v8f acc[4] = {};
  for (int k0 = 0; k0 < F; k0 += 32) {
    // A fragment 16x32 bf16: K chunks [hi*8, hi*8+8) and [16+hi*8, ...)
    float fa[16];
    const float* ap = hrow + k0 + (hi << 3);
    *(float4*)&fa[0]  = *(const float4*)(ap);
    *(float4*)&fa[4]  = *(const float4*)(ap + 4);
    *(float4*)&fa[8]  = *(const float4*)(ap + 16);
    *(float4*)&fa[12] = *(const float4*)(ap + 20);
    v16bf aF;
#pragma unroll
    for (int e = 0; e < 16; ++e) aF[e] = (bf16_t)fa[e];
#pragma unroll
    for (int s = 0; s < 4; ++s) {
      const int n = n0 + s * 16 + lr;    // B column in this lane
      const bf16_t* bp = Wt + (size_t)n * F + k0 + (hi << 3);
      v16bf bF = mk16(*(const v8bf*)bp, *(const v8bf*)(bp + 16));
      acc[s] = __builtin_amdgcn_wmma_f32_16x16x32_bf16(
          false, aF, false, bF, (short)0, acc[s], false, false);
    }
  }
  // C/D layout: VGPR v holds row (i0 + v + 8*hi), col = n0 + s*16 + lr.
  // WhbfT gets 8 consecutive row-positions per lane -> one b128 store.
  const int rbase = i0 + (hi << 3);
#pragma unroll
  for (int s = 0; s < 4; ++s) {
    const int col = n0 + s * 16 + lr;
    v8bf wv;
#pragma unroll
    for (int v = 0; v < 8; ++v) {
      bf16_t w = (bf16_t)acc[s][v];
      wv[v] = w;
      Whbf[((size_t)b * N2 + rbase + v) * F + col] = w;   // row-major copy
    }
    *(v8bf*)&WhbfT[((size_t)b * F + col) * N2 + rbase] = wv;  // [n][j] copy
  }
}

// ---- Kernel 2: s1 = Wh . a[:F],  s2 = Wh . a[F:]  (one wave per row) -----
__global__ __launch_bounds__(256) void k_sdots(const bf16_t* __restrict__ Whbf,
                                               const float* __restrict__ avec,
                                               float* __restrict__ s1,
                                               float* __restrict__ s2) {
  const int wave = threadIdx.x >> 5;
  const int lane = threadIdx.x & 31;
  const int row  = blockIdx.x * 8 + wave;          // global row
  const bf16_t* wp = Whbf + (size_t)row * F + lane * 16;
  v8bf w0 = *(const v8bf*)wp;
  v8bf w1 = *(const v8bf*)(wp + 8);
  float acc1 = 0.f, acc2 = 0.f;
  const float* a1 = avec + lane * 16;
  const float* a2 = avec + F + lane * 16;
#pragma unroll
  for (int e = 0; e < 8; ++e) {
    float f0 = (float)w0[e], f1 = (float)w1[e];
    acc1 += f0 * a1[e] + f1 * a1[e + 8];
    acc2 += f0 * a2[e] + f1 * a2[e + 8];
  }
#pragma unroll
  for (int off = 16; off > 0; off >>= 1) {
    acc1 += __shfl_xor(acc1, off, 32);
    acc2 += __shfl_xor(acc2, off, 32);
  }
  if (lane == 0) { s1[row] = acc1; s2[row] = acc2; }
}

// ---- Kernel 3: softmax stats pass (row max m, denom l), streaming adj ----
__global__ __launch_bounds__(256) void k_stats(const int* __restrict__ adj,
                                               const float* __restrict__ s1,
                                               const float* __restrict__ s2,
                                               float* __restrict__ mrow,
                                               float* __restrict__ lrow) {
  const int wave = threadIdx.x >> 5;
  const int lane = threadIdx.x & 31;
  const int row  = blockIdx.x * 8 + wave;
  const int b = row >> 11;
  const float s1i = s1[row];
  const float* s2b = s2 + (size_t)b * N2;
  const int* arow = adj + (size_t)row * (size_t)N2;
  float m = -3.0e38f, l = 0.f;
  for (int j = lane * 4; j < N2; j += 128) {
    int4   av  = *(const int4*)(arow + j);
    float4 s2v = *(const float4*)(s2b + j);
    if (j + 512 < N2) __builtin_prefetch(arow + j + 512, 0, 1);
    float e[4] = {s1i + s2v.x, s1i + s2v.y, s1i + s2v.z, s1i + s2v.w};
    int   msk[4] = {av.x, av.y, av.z, av.w};
#pragma unroll
    for (int q = 0; q < 4; ++q) {
      float ev = (e[q] > 0.f) ? e[q] : GAT_ALPHA * e[q];
      if (msk[q] <= 0) ev = GAT_NEG_INF;
      if (ev > m) { l *= __expf(m - ev); m = ev; }
      l += __expf(ev - m);
    }
  }
#pragma unroll
  for (int off = 16; off > 0; off >>= 1) {
    float mo = __shfl_xor(m, off, 32);
    float lo = __shfl_xor(l, off, 32);
    float mn = fmaxf(m, mo);
    l = l * __expf(m - mn) + lo * __expf(mo - mn);
    m = mn;
  }
  if (lane == 0) { mrow[row] = m; lrow[row] = l; }
}

// ---- Kernel 4: h' = elu(softmax(e) @ Wh) ------------------------------
// 16-row tile per block; 16x64 P tile built cooperatively in LDS, double-
// buffered (one barrier per 64-wide k-step); wave w does out-cols [64w,64w+64).
__global__ __launch_bounds__(256) void k_attn_pv(const int* __restrict__ adj,
                                                 const float* __restrict__ s1,
                                                 const float* __restrict__ s2,
                                                 const float* __restrict__ mrow,
                                                 const float* __restrict__ lrow,
                                                 const bf16_t* __restrict__ WhbfT,
                                                 float* __restrict__ out) {
  __shared__ bf16_t pT[2][16 * 64];     // 2 x 2KB double-buffered P tile
  const int rt   = blockIdx.x;
  const int tid  = threadIdx.x;
  const int wave = tid >> 5;
  const int lane = tid & 31;
  const int b    = rt >> 7;
  const int i0   = (rt & 127) << 4;
  // P-producer role: thread handles (row = tid>>4, 4 adjacent j columns)
  const int prow = tid >> 4;            // 0..15
  const int jc   = (tid & 15) * 4;      // 0..60
  const size_t growP = (size_t)b * N2 + i0 + prow;
  const float s1i = s1[growP];
  const float mi  = mrow[growP];
  const int* arow = adj + growP * (size_t)N2;
  const float* s2b = s2 + (size_t)b * N2;
  // WMMA-consumer role
  const int lr = lane & 15;
  const int hi = lane >> 4;
  const int n0 = wave * 64;

  v8f acc[4] = {};
  for (int j0 = 0, it = 0; j0 < N2; j0 += 64, ++it) {
    const int buf = it & 1;
    // produce 4 unnormalized P values into the LDS tile
    int4   av  = *(const int4*)(arow + j0 + jc);
    float4 s2v = *(const float4*)(s2b + j0 + jc);
    if (j0 + 256 < N2) __builtin_prefetch(arow + j0 + 256 + jc, 0, 1);
    float e[4] = {s1i + s2v.x, s1i + s2v.y, s1i + s2v.z, s1i + s2v.w};
    int   msk[4] = {av.x, av.y, av.z, av.w};
    v4bf pv;
#pragma unroll
    for (int q = 0; q < 4; ++q) {
      float ev = (e[q] > 0.f) ? e[q] : GAT_ALPHA * e[q];
      if (msk[q] <= 0) ev = GAT_NEG_INF;
      pv[q] = (bf16_t)__expf(ev - mi);
    }
    *(v4bf*)&pT[buf][prow * 64 + jc] = pv;
    __syncthreads();   // waits DScnt: also fences reads of this buf 2 iters ago
    // A fragments (16x32 each) from LDS: ds_load_b128 pairs
    const bf16_t* app = &pT[buf][lr * 64];
    v16bf aF0 = mk16(*(const v8bf*)(app + (hi << 3)),
                     *(const v8bf*)(app + 16 + (hi << 3)));
    v16bf aF1 = mk16(*(const v8bf*)(app + 32 + (hi << 3)),
                     *(const v8bf*)(app + 48 + (hi << 3)));
#pragma unroll
    for (int s = 0; s < 4; ++s) {
      const int n = n0 + s * 16 + lr;
      const bf16_t* bp = WhbfT + ((size_t)b * F + n) * (size_t)N2 + j0 + (hi << 3);
      v16bf bF0 = mk16(*(const v8bf*)bp,        *(const v8bf*)(bp + 16));
      v16bf bF1 = mk16(*(const v8bf*)(bp + 32), *(const v8bf*)(bp + 48));
      acc[s] = __builtin_amdgcn_wmma_f32_16x16x32_bf16(
          false, aF0, false, bF0, (short)0, acc[s], false, false);
      acc[s] = __builtin_amdgcn_wmma_f32_16x16x32_bf16(
          false, aF1, false, bF1, (short)0, acc[s], false, false);
    }
  }
  // Epilogue: normalize by l, ELU, scatter to concatenated outputs.
  const int rbase = i0 + (hi << 3);
  float linv[8];
#pragma unroll
  for (int v = 0; v < 8; ++v)
    linv[v] = 1.0f / lrow[(size_t)b * N2 + rbase + v];
#pragma unroll
  for (int s = 0; s < 4; ++s) {
    const int col = n0 + s * 16 + lr;
#pragma unroll
    for (int v = 0; v < 8; ++v) {
      const int rp = rbase + v;
      float val = acc[s][v] * linv[v];
      val = (val > 0.f) ? val : (__expf(val) - 1.0f);
      size_t oidx = (rp < T)
          ? ((size_t)b * T + rp) * F + col
          : (size_t)BS * T * F + ((size_t)b * T + (rp - T)) * F + col;
      out[oidx] = val;
    }
  }
}

extern "C" void kernel_launch(void* const* d_in, const int* in_sizes, int n_in,
                              void* d_out, int out_size, void* d_ws, size_t ws_size,
                              hipStream_t stream) {
  (void)in_sizes; (void)n_in; (void)out_size; (void)ws_size;
  const float* xa  = (const float*)d_in[0];
  const float* xv  = (const float*)d_in[1];
  const int*   adj = (const int*)d_in[2];
  const float* W   = (const float*)d_in[3];
  const float* a   = (const float*)d_in[4];
  float* out = (float*)d_out;

  const size_t WT_BYTES = (size_t)F * F * sizeof(bf16_t);          // 0.5 MB
  const size_t WH_BYTES = (size_t)BS * N2 * F * sizeof(bf16_t);    // 33.5 MB
  char* wsp = (char*)d_ws;
  bf16_t* Wt    = (bf16_t*)wsp;
  bf16_t* Whbf  = (bf16_t*)(wsp + WT_BYTES);
  bf16_t* WhbfT = (bf16_t*)(wsp + WT_BYTES + WH_BYTES);
  float*  s1    = (float*)(wsp + WT_BYTES + 2 * WH_BYTES);
  float*  s2    = s1 + NROW;
  float*  mr    = s2 + NROW;
  float*  lr    = mr + NROW;

  k_wprep  <<<dim3((F * F) / 256), dim3(256), 0, stream>>>(W, Wt);
  k_gemm_wh<<<dim3(NROW / 16),     dim3(256), 0, stream>>>(xa, xv, Wt, Whbf, WhbfT);
  k_sdots  <<<dim3(NROW / 8),      dim3(256), 0, stream>>>(Whbf, a, s1, s2);
  k_stats  <<<dim3(NROW / 8),      dim3(256), 0, stream>>>(adj, s1, s2, mr, lr);
  k_attn_pv<<<dim3(NROW / 16),     dim3(256), 0, stream>>>(adj, s1, s2, mr, lr, WhbfT, out);
}
